// LIFGatedRouter_18717467476425
// MI455X (gfx1250) — compile-verified, matrix operand order checked by reading, not run on hardware
//
#include <hip/hip_runtime.h>
#include <cstdint>

// ---------------------------------------------------------------------------
// LIFGatedRouter on MI455X (gfx1250):
//   pass 0: split x, Wg, Wu, Wd into bf16 hi/lo planes (streaming, once)
//   G = x*Wg^T ; U = x*Wu^T        (bf16x3 split WMMA GEMM, f32 accumulate)
//   LIF scan over T -> H = spike * silu(U)  (emitted as bf16 hi/lo planes)
//   out = H*Wd^T                   (same WMMA GEMM)
// Shapes: B=4 T=2048 D=1024 F=4096 ; BT = 8192.
// All operand conversion is hoisted out of the GEMM inner loop: fragments are
// straight ds_load_b128 of pre-split bf16 planes (no per-K-step VALU storm).
// ---------------------------------------------------------------------------

#define MBLK 128   // block tile rows (M)
#define NBLK 64    // block tile cols (N)
#define KSTEP 32   // K elements per stage (matches 16x16x32 bf16 WMMA)
#define KPADH 40   // padded K stride in LDS (bf16 units; 80B rows, 16B aligned)

typedef __attribute__((ext_vector_type(16))) __bf16 v16bf;
typedef __attribute__((ext_vector_type(8)))  float  v8f;
typedef unsigned short u16;

// ---------------- async global->LDS (guarded; sync fallback) ----------------
#if defined(__has_builtin)
#if __has_builtin(__builtin_amdgcn_global_load_async_to_lds_b128)
#define USE_ASYNC_LDS 1
#endif
#endif

#if defined(USE_ASYNC_LDS)
// Builtin takes pointers to 16B int vectors in AS1 (global) / AS3 (LDS).
typedef int v4i_gcc __attribute__((vector_size(16)));
typedef __attribute__((address_space(1))) v4i_gcc* as1_v4i;
typedef __attribute__((address_space(3))) v4i_gcc* as3_v4i;
__device__ __forceinline__ void cp16_to_lds(const void* g, void* l) {
  // generic->as3: low 32 bits of a generic shared address are the LDS offset
  __builtin_amdgcn_global_load_async_to_lds_b128(
      (as1_v4i)(unsigned long long)(uintptr_t)g,
      (as3_v4i)(unsigned int)(uintptr_t)l, 0, 0);
}
#if __has_builtin(__builtin_amdgcn_s_wait_asynccnt)
#define WAIT_ASYNC(n) __builtin_amdgcn_s_wait_asynccnt(n)
#else
#define WAIT_ASYNC(n) asm volatile("s_wait_asynccnt %0" ::"n"(n) : "memory")
#endif
#else
__device__ __forceinline__ void cp16_to_lds(const void* g, void* l) {
  *(uint4*)l = *(const uint4*)g;
}
#define WAIT_ASYNC(n) ((void)0)
#endif

// ---------------- f32 -> bf16 (RNE) split helpers ---------------------------
__device__ __forceinline__ unsigned int bf16_rne(float f) {
  unsigned int x = __float_as_uint(f);
  unsigned int r = 0x7fffu + ((x >> 16) & 1u);
  return (x + r) >> 16;   // low 16 bits valid
}
__device__ __forceinline__ float bf16_to_f32(unsigned int h) {
  return __uint_as_float((h & 0xffffu) << 16);
}

// Split an f32 array into bf16 hi/lo planes: src ~= hi + lo elementwise.
__global__ __launch_bounds__(256) void split_f32_bf16(
    const float* __restrict__ src, u16* __restrict__ hi, u16* __restrict__ lo,
    int n) {
  int i = blockIdx.x * 256 + threadIdx.x;
  if (i >= n) return;
  float f = src[i];
  unsigned int h = bf16_rne(f);
  unsigned int l = bf16_rne(f - bf16_to_f32(h));
  hi[i] = (u16)h;
  lo[i] = (u16)l;
}

union FragU { v16bf v; unsigned int u[8]; uint4 q[2]; };

// One 16x32 bf16 operand fragment straight from LDS (ISA 7.12.2, wave32):
// lane row = lane&15; lanes 0-15 hold K {0..7,16..23}, 16-31 hold {8..15,24..31}.
// p = tile + row*KPADH + kb with kb=(lane>>4)*8 -> two 16B runs, 2x ds_load_b128.
__device__ __forceinline__ void load_frag(FragU& fr, const u16* p) {
  fr.q[0] = *(const uint4*)(p);
  fr.q[1] = *(const uint4*)(p + 16);
}

// Stage bf16 hi/lo tiles: A 128x32, B 64x32 (x2 planes). 6 b128 issues/thread.
__device__ __forceinline__ void stage_copy(
    u16* sAhb, u16* sAlb, u16* sBhb, u16* sBlb, const u16* Ah, const u16* Al,
    const u16* Bh, const u16* Bl, int m0, int n0, int k0, int K, int tid) {
#pragma unroll
  for (int i = 0; i < 2; ++i) {            // A planes: 128 rows x 4 chunks
    int idx = i * 256 + tid;
    int row = idx >> 2, c = (idx & 3) * 8;
    size_t g = (size_t)(m0 + row) * K + k0 + c;
    int l = row * KPADH + c;
    cp16_to_lds(Ah + g, sAhb + l);
    cp16_to_lds(Al + g, sAlb + l);
  }
  {                                        // B planes: 64 rows x 4 chunks
    int row = tid >> 2, c = (tid & 3) * 8;
    size_t g = (size_t)(n0 + row) * K + k0 + c;
    int l = row * KPADH + c;
    cp16_to_lds(Bh + g, sBhb + l);
    cp16_to_lds(Bl + g, sBlb + l);
  }
}

// C[M,N] = A[M,K] * B[N,K]^T from pre-split bf16 planes, bf16x3 accumulation.
// Requires M%128==0, N%64==0, K%32==0 (true for all three GEMMs here).
__global__ __launch_bounds__(256) void gemm_abt_bf16x3(
    const u16* __restrict__ Ah, const u16* __restrict__ Al,
    const u16* __restrict__ Bh, const u16* __restrict__ Bl,
    float* __restrict__ C, int M, int N, int K) {
  __shared__ __align__(16) u16 sAh[2][MBLK * KPADH];  // 2 * 10 KB
  __shared__ __align__(16) u16 sAl[2][MBLK * KPADH];  // 2 * 10 KB
  __shared__ __align__(16) u16 sBh[2][NBLK * KPADH];  // 2 *  5 KB
  __shared__ __align__(16) u16 sBl[2][NBLK * KPADH];  // 2 *  5 KB

  const int tid  = threadIdx.x;
  const int lane = tid & 31;
  const int wave = tid >> 5;
  const int wm   = wave & 3;             // 4 waves along M
  const int wn   = wave >> 2;            // 2 waves along N
  const int m0   = blockIdx.y * MBLK;
  const int n0   = blockIdx.x * NBLK;

  v8f acc[2][2] = {};

  const int nk = K / KSTEP;
  stage_copy(sAh[0], sAl[0], sBh[0], sBl[0], Ah, Al, Bh, Bl, m0, n0, 0, K,
             tid);

  const int kb   = (lane >> 4) * 8;
  const int rl   = lane & 15;
  const int aoff = (wm * 32 + rl) * KPADH + kb;   // mi adds 16*KPADH
  const int boff = (wn * 32 + rl) * KPADH + kb;   // ni adds 16*KPADH

  for (int ks = 0; ks < nk; ++ks) {
    const int cur = ks & 1;
    if (ks + 1 < nk)
      stage_copy(sAh[cur ^ 1], sAl[cur ^ 1], sBh[cur ^ 1], sBl[cur ^ 1], Ah,
                 Al, Bh, Bl, m0, n0, (ks + 1) * KSTEP, K, tid);
    if (ks + 1 < nk) { WAIT_ASYNC(6); } else { WAIT_ASYNC(0); }
    __syncthreads();

    FragU fah[2], fal[2], fbh[2], fbl[2];
#pragma unroll
    for (int mi = 0; mi < 2; ++mi) {
      load_frag(fah[mi], &sAh[cur][aoff + mi * 16 * KPADH]);
      load_frag(fal[mi], &sAl[cur][aoff + mi * 16 * KPADH]);
    }
#pragma unroll
    for (int ni = 0; ni < 2; ++ni) {
      load_frag(fbh[ni], &sBh[cur][boff + ni * 16 * KPADH]);
      load_frag(fbl[ni], &sBl[cur][boff + ni * 16 * KPADH]);
    }

#pragma unroll
    for (int mi = 0; mi < 2; ++mi)
#pragma unroll
      for (int ni = 0; ni < 2; ++ni) {
        acc[mi][ni] = __builtin_amdgcn_wmma_f32_16x16x32_bf16(
            false, fah[mi].v, false, fbh[ni].v, (short)0, acc[mi][ni], false,
            false);
        acc[mi][ni] = __builtin_amdgcn_wmma_f32_16x16x32_bf16(
            false, fah[mi].v, false, fbl[ni].v, (short)0, acc[mi][ni], false,
            false);
        acc[mi][ni] = __builtin_amdgcn_wmma_f32_16x16x32_bf16(
            false, fal[mi].v, false, fbh[ni].v, (short)0, acc[mi][ni], false,
            false);
      }
    __syncthreads();
  }

  // Epilogue: 16x16 f32 C layout — VGPR r holds M=r (lanes 0-15) / r+8 (16-31)
  const int lhalf = lane & 15;
  const int lsel  = lane >> 4;
#pragma unroll
  for (int mi = 0; mi < 2; ++mi)
#pragma unroll
    for (int ni = 0; ni < 2; ++ni) {
      int row0 = m0 + wm * 32 + mi * 16 + lsel * 8;
      int col  = n0 + wn * 32 + ni * 16 + lhalf;
#pragma unroll
      for (int r = 0; r < 8; ++r)
        C[(size_t)(row0 + r) * N + col] = acc[mi][ni][r];
    }
}

// ---------------- LIF scan + silu gating (sequential in T) ------------------
// One thread per (b,f); threads consecutive in f -> coalesced per timestep.
// Emits H = spike * silu(U) directly as bf16 hi/lo planes for GEMM-3.
__global__ __launch_bounds__(256) void lif_gate_silu(
    const float* __restrict__ G, const float* __restrict__ U,
    const float* __restrict__ theta, u16* __restrict__ Hhi,
    u16* __restrict__ Hlo, int B, int T, int F) {
  int idx = blockIdx.x * blockDim.x + threadIdx.x;
  if (idx >= B * F) return;
  int b = idx / F, f = idx % F;
  float m = 0.0f;
  float th = theta[f];
  size_t off = (size_t)b * T * F + f;
  for (int t = 0; t < T; ++t, off += F) {
    m = 0.8f * m + G[off];                     // leaky integrate (BETA=0.8)
    float spike = (m >= 1.0f) ? 1.0f : 0.0f;   // threshold fire
    m -= th * spike;                           // reset
    float u = U[off];
    float h = spike * (u / (1.0f + __expf(-u)));  // gate * silu(up)
    unsigned int hh = bf16_rne(h);
    unsigned int hl = bf16_rne(h - bf16_to_f32(hh));
    Hhi[off] = (u16)hh;
    Hlo[off] = (u16)hl;
  }
}

// ---------------------------------------------------------------------------
extern "C" void kernel_launch(void* const* d_in, const int* in_sizes, int n_in,
                              void* d_out, int out_size, void* d_ws,
                              size_t ws_size, hipStream_t stream) {
  const int Bb = 4, T = 2048, D = 1024, F = 4096;
  const int BT = Bb * T;                 // 8192
  const size_t nX = (size_t)BT * D;      // 8.39M
  const size_t nW = (size_t)F * D;       // 4.19M (Wg, Wu, Wd all F*D)
  const size_t nG = (size_t)BT * F;      // 33.6M

  const float* x     = (const float*)d_in[0];
  const float* Wg    = (const float*)d_in[1];
  const float* Wu    = (const float*)d_in[2];
  const float* Wd    = (const float*)d_in[3];
  const float* theta = (const float*)d_in[4];
  float* out = (float*)d_out;

  // Workspace carve-up (bytes): f32 G,U then bf16 planes.
  char* w = (char*)d_ws;
  float* G   = (float*)w;            w += nG * 4;   // 134 MB
  float* U   = (float*)w;            w += nG * 4;   // 134 MB
  u16* Hhi   = (u16*)w;              w += nG * 2;   //  67 MB
  u16* Hlo   = (u16*)w;              w += nG * 2;   //  67 MB
  u16* xhi   = (u16*)w;              w += nX * 2;
  u16* xlo   = (u16*)w;              w += nX * 2;
  u16* Wghi  = (u16*)w;              w += nW * 2;
  u16* Wglo  = (u16*)w;              w += nW * 2;
  u16* Wuhi  = (u16*)w;              w += nW * 2;
  u16* Wulo  = (u16*)w;              w += nW * 2;
  u16* Wdhi  = (u16*)w;              w += nW * 2;
  u16* Wdlo  = (u16*)w;              w += nW * 2;

  dim3 blk(256);

  // Pass 0: split operands into bf16 hi/lo planes (streaming, tiny cost).
  split_f32_bf16<<<dim3((nX + 255) / 256), blk, 0, stream>>>(x, xhi, xlo,
                                                             (int)nX);
  split_f32_bf16<<<dim3((nW + 255) / 256), blk, 0, stream>>>(Wg, Wghi, Wglo,
                                                             (int)nW);
  split_f32_bf16<<<dim3((nW + 255) / 256), blk, 0, stream>>>(Wu, Wuhi, Wulo,
                                                             (int)nW);
  split_f32_bf16<<<dim3((nW + 255) / 256), blk, 0, stream>>>(Wd, Wdhi, Wdlo,
                                                             (int)nW);

  // G = x*Wg^T ; U = x*Wu^T
  dim3 g1(F / NBLK, BT / MBLK);          // 64 x 64 blocks
  gemm_abt_bf16x3<<<g1, blk, 0, stream>>>(xhi, xlo, Wghi, Wglo, G, BT, F, D);
  gemm_abt_bf16x3<<<g1, blk, 0, stream>>>(xhi, xlo, Wuhi, Wulo, U, BT, F, D);

  // LIF scan + gating
  lif_gate_silu<<<dim3((Bb * F) / 256), blk, 0, stream>>>(G, U, theta, Hhi,
                                                          Hlo, Bb, T, F);

  // out = H*Wd^T
  dim3 g3(D / NBLK, BT / MBLK);          // 16 x 64 blocks
  gemm_abt_bf16x3<<<g3, blk, 0, stream>>>(Hhi, Hlo, Wdhi, Wdlo, out, BT, D,
                                          F);
}
